// HungarianMatcherDynamicK_84155589198425
// MI455X (gfx1250) — matrix operand order repaired and checked by reading, not compiled, not run on hardware
//
#include <hip/hip_runtime.h>
#include <hip/hip_bf16.h>
#include <stdint.h>

// ---------------- problem constants ----------------
constexpr int   kB = 16, kQ = 2000, kC = 80, kG = 100;
constexpr int   kKPad = 96;          // C padded to 3 chunks of 32 for WMMA K
constexpr int   kNT   = 7;           // ceil(G/16) N-tiles
constexpr int   kQT   = kQ / 16;     // 125 M-tiles
constexpr int   kOtaK = 5, kMaxIters = 1000;
constexpr float kAlpha = 0.25f, kCostClass = 2.0f, kCostBbox = 5.0f, kCostGiou = 2.0f, kRad = 2.5f;
// d_out layout (floats): selected[B*Q] | gt_idx[B*Q] | matched_qid[B*G] | M[B*Q*G]
constexpr size_t OFF_SEL = 0;
constexpr size_t OFF_GT  = (size_t)kB * kQ;
constexpr size_t OFF_MQ  = (size_t)2 * kB * kQ;
constexpr size_t OFF_M   = (size_t)2 * kB * kQ + (size_t)kB * kG;

typedef __attribute__((ext_vector_type(16))) _Float16 v16h;
typedef __attribute__((ext_vector_type(8)))  float    v8f;
typedef int v4i_raw __attribute__((vector_size(16))); // matches builtin's pointee type

// ---------------- CDNA5 async LDS staging (guarded) ----------------
#if defined(__AMDGCN__) && __has_builtin(__builtin_amdgcn_global_load_async_to_lds_b128)
#define ASYNC_OK 1
#else
#define ASYNC_OK 0
#endif

__device__ __forceinline__ void copy16_to_lds(const void* g, void* l) {
#if ASYNC_OK
  __builtin_amdgcn_global_load_async_to_lds_b128(
      (__attribute__((address_space(1))) v4i_raw*)(uintptr_t)g,
      (__attribute__((address_space(3))) v4i_raw*)l, 0, 0);
#else
  *(float4*)l = *(const float4*)g;
#endif
}

__device__ __forceinline__ void async_wait_all() {
#if ASYNC_OK
#if __has_builtin(__builtin_amdgcn_s_wait_asynccnt)
  __builtin_amdgcn_s_wait_asynccnt(0);
#else
  asm volatile("s_wait_asynccnt 0" ::: "memory");
#endif
#endif
}

__device__ __forceinline__ float focal_diff(float l) {
  float p   = 1.0f / (1.0f + __expf(-l));
  float pos = kAlpha * (1.0f - p) * (1.0f - p) * (-__logf(p + 1e-8f));
  float neg = (1.0f - kAlpha) * p * p * (-__logf(1.0f - p + 1e-8f));
  return pos - neg;
}

// ==================================================================
// Kernel 1: cost + iou matrices, layout [b][g][q].
// 256 threads = 8 waves; each wave owns one 16-query WMMA row-tile.
// cost_class via v_wmma_f32_16x16x32_f16 (gather-as-GEMM with one-hot B).
// ==================================================================
__global__ __launch_bounds__(256) void cost_kernel(
    const float* __restrict__ logits,  // [B][Q][C]
    const float* __restrict__ pboxes,  // [B][Q][4] cxcywh
    const float* __restrict__ gboxes,  // [B][G][4] cxcywh
    const int*   __restrict__ labels,  // [B][G]
    const float* __restrict__ img,     // [B][4]
    float* __restrict__ costO,         // [B][G][Q]
    float* __restrict__ iouO)          // [B][G][Q]
{
  const int bidx  = blockIdx.y;
  const int tid   = threadIdx.x;
  const int qbase = blockIdx.x * 128;   // 8 qtiles per block

  __shared__ __align__(16) float sRaw[kG * 4];
  __shared__ __align__(16) int   sLab[kG];
  __shared__ __align__(16) float sImg[4];
  __shared__ float sGxy[kG * 4];   // gt abs xyxy
  __shared__ float sGc [kG * 4];   // gt abs cxcywh
  __shared__ float sGn [kG * 4];   // gt normalized xyxy
  __shared__ float sGarea[kG];
  __shared__ float sQxy[128 * 4];  // query abs xyxy
  __shared__ float sQn [128 * 4];  // query normalized xyxy
  __shared__ unsigned char sFg[128];

  // ---- stage gt boxes / labels / img via async 16B transfers ----
  if (tid < kG) {
    copy16_to_lds(gboxes + ((size_t)bidx * kG + tid) * 4, &sRaw[tid * 4]);
  } else if (tid < kG + 25) {
    int j = tid - kG;
    copy16_to_lds(labels + (size_t)bidx * kG + j * 4, &sLab[j * 4]);
  } else if (tid == kG + 25) {
    copy16_to_lds(img + (size_t)bidx * 4, &sImg[0]);
  }
  async_wait_all();
  __syncthreads();

  // ---- derived gt quantities ----
  if (tid < kG) {
    float cx = sRaw[tid*4+0], cy = sRaw[tid*4+1], w = sRaw[tid*4+2], h = sRaw[tid*4+3];
    float n0 = cx - 0.5f*w, n1 = cy - 0.5f*h, n2 = cx + 0.5f*w, n3 = cy + 0.5f*h;
    float x1 = n0 * sImg[0], y1 = n1 * sImg[1], x2 = n2 * sImg[2], y2 = n3 * sImg[3];
    sGxy[tid*4+0]=x1; sGxy[tid*4+1]=y1; sGxy[tid*4+2]=x2; sGxy[tid*4+3]=y2;
    sGc [tid*4+0]=(x1+x2)*0.5f; sGc[tid*4+1]=(y1+y2)*0.5f; sGc[tid*4+2]=x2-x1; sGc[tid*4+3]=y2-y1;
    sGn [tid*4+0]=n0; sGn[tid*4+1]=n1; sGn[tid*4+2]=n2; sGn[tid*4+3]=n3;
    sGarea[tid] = (x2-x1)*(y2-y1);
  }
  __syncthreads();

  // ---- per-query box prep + fg_mask (any in_box | any in_ctr over all gts) ----
  if (tid < 128) {
    int q = qbase + tid;
    unsigned char fg = 0;
    if (q < kQ) {
      const float* pb = pboxes + ((size_t)bidx * kQ + q) * 4;
      float cx = pb[0], cy = pb[1], w = pb[2], h = pb[3];
      float n0 = cx - 0.5f*w, n1 = cy - 0.5f*h, n2 = cx + 0.5f*w, n3 = cy + 0.5f*h;
      float a0 = n0*sImg[0], a1 = n1*sImg[1], a2 = n2*sImg[2], a3 = n3*sImg[3];
      sQn [tid*4+0]=n0; sQn[tid*4+1]=n1; sQn[tid*4+2]=n2; sQn[tid*4+3]=n3;
      sQxy[tid*4+0]=a0; sQxy[tid*4+1]=a1; sQxy[tid*4+2]=a2; sQxy[tid*4+3]=a3;
      float ax = (a0+a2)*0.5f, ay = (a1+a3)*0.5f;
      for (int g = 0; g < kG; ++g) {
        float gx1=sGxy[g*4+0], gy1=sGxy[g*4+1], gx2=sGxy[g*4+2], gy2=sGxy[g*4+3];
        float gcx=sGc[g*4+0], gcy=sGc[g*4+1], gw=sGc[g*4+2], gh=sGc[g*4+3];
        bool inb = (ax>gx1)&&(ax<gx2)&&(ay>gy1)&&(ay<gy2);
        bool inc = (ax>gcx-kRad*gw)&&(ax<gcx+kRad*gw)&&(ay>gcy-kRad*gh)&&(ay<gcy+kRad*gh);
        fg |= (unsigned char)(inb | inc);
      }
    }
    sFg[tid] = fg;
  }
  __syncthreads();

  // ---- WMMA phase ----
  const int lane = tid & 31;
  const int wave = tid >> 5;
  const int half = lane >> 4;
  const int m    = lane & 15;
  const int qtile = blockIdx.x * 8 + wave;
  if (qtile >= kQT) return;

  const int qm = qtile * 16 + m;
  const float* lrow = logits + ((size_t)bidx * kQ + qm) * kC;

  // A fragment: 16x32 f16 per chunk. Lane layout: k = ((i>>3)<<4) + half*8 + (i&7)
  v16h A[3];
#pragma unroll
  for (int ch = 0; ch < 3; ++ch) {
#pragma unroll
    for (int i = 0; i < 16; ++i) {
      int k = ((i >> 3) << 4) + (half << 3) + (i & 7);
      int c = ch * 32 + k;
      float f = 0.0f;
      if (c < kC) f = focal_diff(lrow[c]);
      A[ch][i] = (_Float16)f;
    }
  }

  for (int nt = 0; nt < kNT; ++nt) {
    int gcol = nt * 16 + m;                 // B column n == C/D column n == lane&15
    int lab  = (gcol < kG) ? sLab[gcol] : -1;
    v8f acc = 0.0f;
#pragma unroll
    for (int ch = 0; ch < 3; ++ch) {
      v16h Bf;                              // B layout: k = ((i>>3)<<4) + half*8 + (i&7)
#pragma unroll
      for (int i = 0; i < 16; ++i) {
        int c = ch * 32 + ((i >> 3) << 4) + (half << 3) + (i & 7);
        Bf[i] = (c == lab) ? (_Float16)1.0f : (_Float16)0.0f;
      }
      acc = __builtin_amdgcn_wmma_f32_16x16x32_f16(false, A[ch], false, Bf,
                                                   (short)0, acc, false, false);
    }
    if (gcol < kG) {
      float gx1=sGxy[gcol*4+0], gy1=sGxy[gcol*4+1], gx2=sGxy[gcol*4+2], gy2=sGxy[gcol*4+3];
      float gcx=sGc[gcol*4+0], gcy=sGc[gcol*4+1], gw=sGc[gcol*4+2], gh=sGc[gcol*4+3];
      float gn0=sGn[gcol*4+0], gn1=sGn[gcol*4+1], gn2=sGn[gcol*4+2], gn3=sGn[gcol*4+3];
      float gar = sGarea[gcol];
#pragma unroll
      for (int r = 0; r < 8; ++r) {
        int ql = wave * 16 + half * 8 + r;  // C/D row: M = r + 8*half
        int q  = qbase + ql;
        float a0=sQxy[ql*4+0], a1=sQxy[ql*4+1], a2=sQxy[ql*4+2], a3=sQxy[ql*4+3];
        float areaA = (a2-a0)*(a3-a1);
        float iw = fmaxf(fminf(a2,gx2)-fmaxf(a0,gx1), 0.0f);
        float ih = fmaxf(fminf(a3,gy2)-fmaxf(a1,gy1), 0.0f);
        float inter = iw*ih;
        float uni = areaA + gar - inter;
        float iou = inter / uni;
        float cw  = fmaxf(fmaxf(a2,gx2)-fminf(a0,gx1), 0.0f);
        float ch2 = fmaxf(fmaxf(a3,gy2)-fminf(a1,gy1), 0.0f);
        float carea = cw*ch2;
        float giou = iou - (carea - uni)/carea;
        float l1 = fabsf(sQn[ql*4+0]-gn0)+fabsf(sQn[ql*4+1]-gn1)
                 + fabsf(sQn[ql*4+2]-gn2)+fabsf(sQn[ql*4+3]-gn3);
        float ax=(a0+a2)*0.5f, ay=(a1+a3)*0.5f;
        bool inb = (ax>gx1)&&(ax<gx2)&&(ay>gy1)&&(ay<gy2);
        bool inc = (ax>gcx-kRad*gw)&&(ax<gcx+kRad*gw)&&(ay>gcy-kRad*gh)&&(ay<gcy+kRad*gh);
        float cst = kCostBbox*l1 + kCostClass*acc[r] - kCostGiou*giou
                  + ((inb&&inc) ? 0.0f : 100.0f) + (sFg[ql] ? 0.0f : 10000.0f);
        size_t o = ((size_t)bidx * kG + gcol) * kQ + q;
        costO[o] = cst;
        iouO[o]  = iou;
      }
    }
  }
}

// ==================================================================
// Kernel 2: dynamic-k matching, one workgroup (32 waves) per batch.
// M kept as per-row 128-bit masks in LDS.
// ==================================================================
__global__ __launch_bounds__(1024) void match_kernel(
    const float* __restrict__ cost,   // [B][G][Q]
    const float* __restrict__ iouS,   // [B][G][Q] (scratch living in d_out M region)
    float* __restrict__ out)
{
  const int b = blockIdx.x;
  const int tid = threadIdx.x;

  __shared__ unsigned       sMask[kQ * 4];
  __shared__ unsigned short sPen[kQ];
  __shared__ unsigned short sArg[kQ];
  __shared__ unsigned char  sC0[kQ];
  __shared__ int sColCnt[kG];
  __shared__ int sDynK[kG];
  __shared__ int sFlags[2];

  for (int q = tid; q < kQ; q += 1024) {
    sMask[q*4+0]=0; sMask[q*4+1]=0; sMask[q*4+2]=0; sMask[q*4+3]=0;
    sPen[q]=0; sC0[q]=0;
  }
  __syncthreads();

  // ---- dyn_k + initial M: per-column top-5 ----
  if (tid < kG) {
    const float* icol = iouS + ((size_t)b * kG + tid) * kQ;
    float tv[kOtaK];
#pragma unroll
    for (int j = 0; j < kOtaK; ++j) tv[j] = -1.0f;
    for (int q = 0; q < kQ; ++q) {
      if ((q & 255) == 0) __builtin_prefetch(icol + q + 512, 0, 1);
      float v = icol[q];
      if (v > tv[kOtaK-1]) {
        int j = kOtaK-1;
        while (j > 0 && v > tv[j-1]) { tv[j] = tv[j-1]; --j; }
        tv[j] = v;
      }
    }
    float s = 0.0f;
#pragma unroll
    for (int j = 0; j < kOtaK; ++j) s += tv[j];
    int dk = (int)s; dk = dk < 1 ? 1 : (dk > kOtaK ? kOtaK : dk);
    sDynK[tid] = dk;

    const float* ccol = cost + ((size_t)b * kG + tid) * kQ;
    float bc[kOtaK]; int bq[kOtaK];
#pragma unroll
    for (int j = 0; j < kOtaK; ++j) { bc[j] = 3.4e38f; bq[j] = 0; }
    for (int q = 0; q < kQ; ++q) {
      if ((q & 255) == 0) __builtin_prefetch(ccol + q + 512, 0, 1);
      float c2 = ccol[q];
      if (c2 < bc[kOtaK-1]) {                 // strict < : stable tie-break
        int j = kOtaK-1;
        while (j > 0 && c2 < bc[j-1]) { bc[j]=bc[j-1]; bq[j]=bq[j-1]; --j; }
        bc[j] = c2; bq[j] = q;
      }
    }
    unsigned bit = 1u << (tid & 31);
    int w = tid >> 5;
    for (int j = 0; j < dk; ++j) atomicOr(&sMask[bq[j]*4 + w], bit);
  }
  __syncthreads();

  // ---- row argmin (penalty-invariant) + conflict0 reset ----
  for (int q = tid; q < kQ; q += 1024) {
    const float* row = cost + (size_t)b * kG * kQ + q;
    float mb = 3.4e38f; int am = 0;
    for (int g = 0; g < kG; ++g) {
      float v = row[(size_t)g * kQ];
      if (v < mb) { mb = v; am = g; }
    }
    sArg[q] = (unsigned short)am;
    int cnt = __popc(sMask[q*4])+__popc(sMask[q*4+1])+__popc(sMask[q*4+2])+__popc(sMask[q*4+3]);
    if (cnt > 1) {
      sC0[q] = 1;
      sMask[q*4+0]=0; sMask[q*4+1]=0; sMask[q*4+2]=0; sMask[q*4+3]=0;
      sMask[q*4 + (am>>5)] = 1u << (am & 31);
    }
  }
  __syncthreads();

  // ---- while loop ----
  int it = 0;
  while (true) {
    if (tid < kG) sColCnt[tid] = 0;
    if (tid == 0) { sFlags[0] = 0; sFlags[1] = 0; }
    __syncthreads();
    for (int q = tid; q < kQ; q += 1024) {
      for (int w = 0; w < 4; ++w) {
        unsigned mw = sMask[q*4+w];
        while (mw) { int bp = __ffs(mw)-1; atomicAdd(&sColCnt[w*32+bp], 1); mw &= mw-1; }
      }
    }
    __syncthreads();
    if (tid < kG && sColCnt[tid] == 0) atomicOr(&sFlags[0], 1);
    __syncthreads();
    if (!(sFlags[0] && it < kMaxIters)) break;

    // cost += 1e5 * matched  (tracked as per-row counters)
    for (int q = tid; q < kQ; q += 1024) {
      int cnt = __popc(sMask[q*4])+__popc(sMask[q*4+1])+__popc(sMask[q*4+2])+__popc(sMask[q*4+3]);
      if (cnt > 0) sPen[q]++;
    }
    __syncthreads();

    // pos = argmin over column (updated cost); add match for unmatched gts
    if (tid < kG && sColCnt[tid] == 0) {
      const float* ccol = cost + ((size_t)b * kG + tid) * kQ;
      float mb = 3.4e38f; int am = 0;
      for (int q = 0; q < kQ; ++q) {
        if ((q & 255) == 0) __builtin_prefetch(ccol + q + 512, 0, 1);
        float v = ccol[q] + 100000.0f * (float)sPen[q];
        if (v < mb) { mb = v; am = q; }
      }
      atomicOr(&sMask[am*4 + (tid>>5)], 1u << (tid & 31));
    }
    __syncthreads();

    for (int q = tid; q < kQ; q += 1024) {
      int cnt = __popc(sMask[q*4])+__popc(sMask[q*4+1])+__popc(sMask[q*4+2])+__popc(sMask[q*4+3]);
      if (cnt > 1) atomicOr(&sFlags[1], 1);
    }
    __syncthreads();
    if (sFlags[1]) {
      for (int q = tid; q < kQ; q += 1024) {
        if (sC0[q]) {
          int am = sArg[q];
          sMask[q*4+0]=0; sMask[q*4+1]=0; sMask[q*4+2]=0; sMask[q*4+3]=0;
          sMask[q*4 + (am>>5)] = 1u << (am & 31);
        }
      }
      __syncthreads();
    }
    ++it;
  }

  // ---- outputs ----
  for (int q = tid; q < kQ; q += 1024) {
    unsigned w0=sMask[q*4], w1=sMask[q*4+1], w2=sMask[q*4+2], w3=sMask[q*4+3];
    int cnt = __popc(w0)+__popc(w1)+__popc(w2)+__popc(w3);
    out[OFF_SEL + (size_t)b*kQ + q] = cnt ? 1.0f : 0.0f;
    int fb = 0;
    if      (w0) fb =      __ffs(w0)-1;
    else if (w1) fb = 32 + __ffs(w1)-1;
    else if (w2) fb = 64 + __ffs(w2)-1;
    else if (w3) fb = 96 + __ffs(w3)-1;
    out[OFF_GT + (size_t)b*kQ + q] = (float)fb;
  }
  if (tid < kG) {
    const float* ccol = cost + ((size_t)b * kG + tid) * kQ;
    unsigned bit = 1u << (tid & 31); int w = tid >> 5;
    float mb = 3.4e38f; int am = 0;
    for (int q = 0; q < kQ; ++q) {
      bool set = (sMask[q*4 + w] & bit) != 0;
      float v = set ? (ccol[q] + 100000.0f * (float)sPen[q]) : 1e30f;
      if (v < mb) { mb = v; am = q; }
    }
    out[OFF_MQ + (size_t)b*kG + tid] = (float)am;
  }
  __syncthreads();   // all iouS/cost reads done; safe to overwrite M region
  for (int idx = tid; idx < kQ * kG; idx += 1024) {
    int q = idx / kG, g = idx % kG;
    bool set = (sMask[q*4 + (g>>5)] >> (g & 31)) & 1u;
    out[OFF_M + (size_t)b*kQ*kG + (size_t)q*kG + g] = set ? 1.0f : 0.0f;
  }
}

// ==================================================================
extern "C" void kernel_launch(void* const* d_in, const int* in_sizes, int n_in,
                              void* d_out, int out_size, void* d_ws, size_t ws_size,
                              hipStream_t stream) {
  (void)in_sizes; (void)n_in; (void)out_size; (void)ws_size;
  const float* logits = (const float*)d_in[0];   // [B][Q][C]
  const float* pboxes = (const float*)d_in[1];   // [B][Q][4]
  const float* gboxes = (const float*)d_in[2];   // [B][G][4]
  const int*   labels = (const int*)  d_in[3];   // [B][G]
  const float* img    = (const float*)d_in[4];   // [B][4]
  float* out = (float*)d_out;
  float* cost_ws = (float*)d_ws;                 // [B][G][Q] floats (12.8 MB)
  float* iou_scratch = out + OFF_M;              // reuse M region as [B][G][Q] iou scratch

  dim3 g1((kQT + 7) / 8, kB);                    // 16 x 16 blocks, 256 threads (8 waves)
  cost_kernel<<<g1, 256, 0, stream>>>(logits, pboxes, gboxes, labels, img,
                                      cost_ws, iou_scratch);
  match_kernel<<<kB, 1024, 0, stream>>>(cost_ws, iou_scratch, out);
}
